// _CrossAttentionBlock_49125835931818
// MI455X (gfx1250) — matrix operand order repaired and checked by reading, not compiled, not run on hardware
//
#include <hip/hip_runtime.h>
#include <hip/hip_bf16.h>
#include <math.h>

// CDNA5 / gfx1250 fused CrossAttentionBlock.
// softmax over a single key == 1.0  =>  attn == v; Wq/Wk are dead.
// Fused GEMM chain with v_wmma_f32_16x16x32_bf16 (f32 accumulate), all
// intermediates in LDS. Weights pre-packed in WMMA-B fragment order (bf16)
// and staged per-K-panel into LDS via the Tensor Data Mover (double-buffered,
// prefetched one panel ahead; s_wait_tensorcnt + workgroup barrier).

typedef __bf16 bf16_t;
typedef __attribute__((ext_vector_type(16))) __bf16 v16bf;
typedef __attribute__((ext_vector_type(8)))  __bf16 v8bf;
typedef __attribute__((ext_vector_type(8)))  float  v8f;
typedef __attribute__((ext_vector_type(4)))  unsigned int v4u;
typedef __attribute__((ext_vector_type(4)))  int v4i;
typedef __attribute__((ext_vector_type(8)))  int v8i;

#define BB   65536
#define DD   512
#define DPP  256
#define BM   64        // rows per workgroup
#define EPSV 1e-5f

// ---- LDS layout (dynamic shared; 265216 B < 320 KB/WGP) ----
// R0 (132096 B): P(bf16 64x264) -> ATTN(bf16 64x520) -> H(bf16 64x1032)
// R1 ( 66560 B): KV(bf16 64x520) -> X(bf16 64x520)
// red (1024 B): per-row LN partials [64][2] sum + [64][2] sumsq
// PNL (2 x 32768 B): double-buffered weight K-panels (TDM destination)
#define SP 264
#define SD 520
#define SH 1032
#define R0_OFF 0
#define R1_OFF 132096
#define RED_OFF (132096 + 66560)
#define PNL_OFF (RED_OFF + 1024)
#define PANEL_BF 16384              // bf16 elems per panel buffer (32 KB)
#define SMEM_BYTES (PNL_OFF + 2 * 32768)

// ---- packed (B-fragment-ordered) bf16 weights in d_ws (bf16 units) ----
// Fragment layout per 16x16 n-tile: 512 bf16 = [half c=0|c=1][lane 0..31][8 bf16],
// halves 256 bf16 apart -> conflict-free ds_load_b128 (4-dword lane stride).
// A K-panel (one kt, all n-tiles) is a contiguous 32 KB block -> 1 TDM copy.
#define WP_OFF 0
#define WV_OFF 131072
#define WO_OFF (131072 + 262144)
#define W1_OFF (131072 + 262144 + 262144)
#define W2_OFF (131072 + 262144 + 262144 + 524288)

// Pack W (N rows x K cols f32, used as B = W^T) into split-half fragment order:
// dword idx -> j(2b) | lane(5b) | c(1b) | (kt*NT+nt); pair k0 = kt*32 + (lane>>4)*16 + c*8 + 2j.
__global__ void pack_w(const float* __restrict__ W, bf16_t* __restrict__ out, int N, int K) {
  int pairs = (N * K) >> 1;
  for (int idx = blockIdx.x * blockDim.x + threadIdx.x; idx < pairs;
       idx += gridDim.x * blockDim.x) {
    int j    = idx & 3;
    int lane = (idx >> 2) & 31;
    int c    = (idx >> 7) & 1;
    int tnk  = idx >> 8;
    int NT   = N >> 4;
    int nt   = tnk % NT;
    int kt   = tnk / NT;
    int n    = (nt << 4) + (lane & 15);
    int k0   = (kt << 5) + ((lane >> 4) << 4) + (c << 3) + (j << 1);
    out[idx * 2 + 0] = (bf16_t)W[(size_t)n * K + k0];
    out[idx * 2 + 1] = (bf16_t)W[(size_t)n * K + k0 + 1];
  }
}

// Tensor Data Mover: 2D copy (rows x row_elems8 of 8-byte elems, global row
// stride row_stride8) from global to LDS (rows land contiguously in LDS).
__device__ __forceinline__ void tdm_copy_2d(unsigned lds_off, unsigned long long gaddr,
                                            unsigned row_elems8, unsigned rows,
                                            unsigned row_stride8) {
  v4u g0;
  g0.x = 1u;                                                // count=1, user mode
  g0.y = lds_off;                                           // LDS byte address
  g0.z = (unsigned)(gaddr & 0xFFFFFFFFu);                   // global_addr lo
  g0.w = (unsigned)((gaddr >> 32) & 0x1FFFFFFu) | (2u << 30); // hi + type=2
  v8i g1;
  g1[0] = (int)(3u << 16);                                  // data_size = 8B
  g1[1] = (int)((row_elems8 & 0xFFFFu) << 16);              // tensor_dim0 lo16
  g1[2] = (int)(((row_elems8 >> 16) & 0xFFFFu) | ((rows & 0xFFFFu) << 16)); // td0 hi | td1 lo
  g1[3] = (int)(((rows >> 16) & 0xFFFFu) | ((row_elems8 & 0xFFFFu) << 16)); // td1 hi | tile_dim0
  g1[4] = (int)(rows & 0xFFFFu);                            // tile_dim1 (tile_dim2=0)
  g1[5] = (int)row_stride8;                                 // tensor_dim0_stride lo32
  g1[6] = 0;                                                // stride0 hi | stride1 lo
  g1[7] = 0;
  v4i z4 = {0, 0, 0, 0};
#if defined(__clang_major__) && (__clang_major__ >= 23)
  v8i z8 = {0, 0, 0, 0, 0, 0, 0, 0};
  __builtin_amdgcn_tensor_load_to_lds(g0, g1, z4, z4, z8, 0);
#else
  __builtin_amdgcn_tensor_load_to_lds(g0, g1, z4, z4, 0);
#endif
}

// One 16-row strip x 16 column-tiles, K = KT*32, A from LDS, B from TDM-staged
// LDS panels (double-buffered, one panel prefetched ahead by wave 0).
template<int KT, int KSTRIDE_B, int ROWSTRIDE8>
__device__ __forceinline__ void gemm_staged(v8f acc[16], const bf16_t* aBase, int astride,
                                            unsigned long long wgaddr, bf16_t* pnl0,
                                            int h, int lane, int wid) {
  const int m = lane & 15, khalf = lane >> 4;
  const int h16 = h << 4;
  if (wid == 0)
    tdm_copy_2d((unsigned)(size_t)(void*)pnl0, wgaddr, 2048u, 2u, (unsigned)ROWSTRIDE8);
  for (int kt = 0; kt < KT; ++kt) {
    if (wid == 0) __builtin_amdgcn_s_wait_tensorcnt(0);   // panel kt landed
    __syncthreads();                                      // publish panel kt
    if (wid == 0 && (kt + 1) < KT) {                      // prefetch panel kt+1
      bf16_t* nb = pnl0 + ((kt + 1) & 1) * PANEL_BF;
      tdm_copy_2d((unsigned)(size_t)(void*)nb,
                  wgaddr + (unsigned long long)(kt + 1) * (unsigned)KSTRIDE_B,
                  2048u, 2u, (unsigned)ROWSTRIDE8);
    }
    const bf16_t* panel = pnl0 + (kt & 1) * PANEL_BF;
    // A-frag (16x32 bf16): lane m holds K = {khalf*16+0..15}
    const bf16_t* arow = aBase + m * astride + kt * 32;
    v8bf alo = *(const v8bf*)(arow + khalf * 16);
    v8bf ahi = *(const v8bf*)(arow + khalf * 16 + 8);
    v16bf a = __builtin_shufflevector(alo, ahi, 0,1,2,3,4,5,6,7,8,9,10,11,12,13,14,15);
#pragma unroll
    for (int t = 0; t < 16; ++t) {
      const bf16_t* pb = panel + (((h16 + t) << 9) + (lane << 3));
      v8bf blo = *(const v8bf*)pb;          // conflict-free: 4-dword lane stride
      v8bf bhi = *(const v8bf*)(pb + 256);
      v16bf b = __builtin_shufflevector(blo, bhi, 0,1,2,3,4,5,6,7,8,9,10,11,12,13,14,15);
      acc[t] = __builtin_amdgcn_wmma_f32_16x16x32_bf16(false, a, false, b,
                                                       (short)0, acc[t], false, false);
    }
  }
}

__global__ void __launch_bounds__(256, 1)
fused_block(const float* __restrict__ cell, const float* __restrict__ pert,
            const bf16_t* __restrict__ wpk,
            const float* __restrict__ bp, const float* __restrict__ bv,
            const float* __restrict__ bo, const float* __restrict__ b1,
            const float* __restrict__ b2,
            const float* __restrict__ g1, const float* __restrict__ be1,
            const float* __restrict__ g2, const float* __restrict__ be2,
            float* __restrict__ out) {
  extern __shared__ char smem[];
  bf16_t* R0  = (bf16_t*)(smem + R0_OFF);
  bf16_t* R1  = (bf16_t*)(smem + R1_OFF);
  float*  red = (float*)(smem + RED_OFF);
  bf16_t* PNL = (bf16_t*)(smem + PNL_OFF);

  const int tid   = threadIdx.x;
  const int lane  = tid & 31;
  const int wid   = tid >> 5;
  const int strip = wid >> 1;   // 0..3 -> rows [strip*16, strip*16+16)
  const int h     = wid & 1;    // column half
  const int n     = lane & 15;
  const int khalf = lane >> 4;
  const int rowbase = blockIdx.x * BM;
  const unsigned long long wpkB = (unsigned long long)(size_t)wpk;

  // ---- Stage A: pert tile -> bf16 P in R0 ----
  for (int i = tid * 4; i < BM * DPP; i += 256 * 4) {
    int r = i >> 8, c = i & 255;
    float4 v = *(const float4*)(pert + (size_t)(rowbase + r) * DPP + c);
    bf16_t* d = R0 + r * SP + c;
    d[0] = (bf16_t)v.x; d[1] = (bf16_t)v.y; d[2] = (bf16_t)v.z; d[3] = (bf16_t)v.w;
  }
  __syncthreads();

  v8f acc[16];

  // ---- Stage B: KV = P @ Wp^T + bp -> R1 ----
  {
    const int col0 = h * 256;
#pragma unroll
    for (int t = 0; t < 16; ++t) {
      float bz = bp[col0 + t * 16 + n];
#pragma unroll
      for (int r = 0; r < 8; ++r) acc[t][r] = bz;
    }
    gemm_staged<8, 32768, 2048>(acc, R0 + strip * 16 * SP, SP,
                                wpkB + (unsigned long long)WP_OFF * 2, PNL, h, lane, wid);
#pragma unroll
    for (int t = 0; t < 16; ++t)
#pragma unroll
      for (int r = 0; r < 8; ++r) {
        int row = strip * 16 + khalf * 8 + r;
        R1[row * SD + col0 + t * 16 + n] = (bf16_t)acc[t][r];
      }
  }
  __syncthreads();

  // ---- Stage C: ATTN = KV @ Wv^T + bv -> R0 (softmax over 1 key == 1) ----
  {
    const int col0 = h * 256;
#pragma unroll
    for (int t = 0; t < 16; ++t) {
      float bz = bv[col0 + t * 16 + n];
#pragma unroll
      for (int r = 0; r < 8; ++r) acc[t][r] = bz;
    }
    gemm_staged<16, 32768, 2048>(acc, R1 + strip * 16 * SD, SD,
                                 wpkB + (unsigned long long)WV_OFF * 2, PNL, h, lane, wid);
#pragma unroll
    for (int t = 0; t < 16; ++t)
#pragma unroll
      for (int r = 0; r < 8; ++r) {
        int row = strip * 16 + khalf * 8 + r;
        R0[row * SD + col0 + t * 16 + n] = (bf16_t)acc[t][r];
      }
  }
  __syncthreads();

  // ---- Stage D: x = LN1(cell + ATTN @ Wo^T + bo) -> bf16 X in R1 ----
  {
    const int col0 = h * 256;
#pragma unroll
    for (int t = 0; t < 16; ++t) {
      float bz = bo[col0 + t * 16 + n];
#pragma unroll
      for (int r = 0; r < 8; ++r) acc[t][r] = bz;
    }
    gemm_staged<16, 32768, 2048>(acc, R0 + strip * 16 * SD, SD,
                                 wpkB + (unsigned long long)WO_OFF * 2, PNL, h, lane, wid);

    float sum[8], sq[8];
#pragma unroll
    for (int r = 0; r < 8; ++r) { sum[r] = 0.f; sq[r] = 0.f; }
#pragma unroll
    for (int t = 0; t < 16; ++t)
#pragma unroll
      for (int r = 0; r < 8; ++r) {
        int grow = rowbase + strip * 16 + khalf * 8 + r;
        float v = acc[t][r] + cell[(size_t)grow * DD + col0 + t * 16 + n];
        acc[t][r] = v; sum[r] += v; sq[r] += v * v;
      }
#pragma unroll
    for (int r = 0; r < 8; ++r)
      for (int msk = 1; msk < 16; msk <<= 1) {
        sum[r] += __shfl_xor(sum[r], msk, 32);
        sq[r]  += __shfl_xor(sq[r],  msk, 32);
      }
    if (n == 0)
#pragma unroll
      for (int r = 0; r < 8; ++r) {
        int row = strip * 16 + khalf * 8 + r;
        red[row * 2 + h] = sum[r];
        red[128 + row * 2 + h] = sq[r];
      }
    __syncthreads();
    float mu[8], rstd[8];
#pragma unroll
    for (int r = 0; r < 8; ++r) {
      int row = strip * 16 + khalf * 8 + r;
      float s = red[row * 2] + red[row * 2 + 1];
      float q = red[128 + row * 2] + red[128 + row * 2 + 1];
      float m_ = s * (1.0f / 512.0f);
      mu[r] = m_;
      rstd[r] = rsqrtf(q * (1.0f / 512.0f) - m_ * m_ + EPSV);
    }
#pragma unroll
    for (int t = 0; t < 16; ++t) {
      int col = col0 + t * 16 + n;
      float gg = g1[col], bb = be1[col];
#pragma unroll
      for (int r = 0; r < 8; ++r) {
        int row = strip * 16 + khalf * 8 + r;
        R1[row * SD + col] = (bf16_t)((acc[t][r] - mu[r]) * rstd[r] * gg + bb);
      }
    }
  }
  __syncthreads();

  // ---- Stage E: H = gelu(X @ W1^T + b1) -> R0 (N=1024: two passes) ----
  for (int np = 0; np < 2; ++np) {
    const int col0 = h * 512 + np * 256;
#pragma unroll
    for (int t = 0; t < 16; ++t) {
      float bz = b1[col0 + t * 16 + n];
#pragma unroll
      for (int r = 0; r < 8; ++r) acc[t][r] = bz;
    }
    // W1 half-panel: two 16 KB chunks (ntiles np*16.. and 32+np*16..) 32 KB apart
    gemm_staged<16, 65536, 4096>(acc, R1 + strip * 16 * SD, SD,
                                 wpkB + (unsigned long long)W1_OFF * 2 + np * 16384ull,
                                 PNL, h, lane, wid);
#pragma unroll
    for (int t = 0; t < 16; ++t)
#pragma unroll
      for (int r = 0; r < 8; ++r) {
        float v = acc[t][r];
        float gel = 0.5f * v * (1.0f + erff(v * 0.70710678118654752f));  // exact GELU
        int row = strip * 16 + khalf * 8 + r;
        R0[row * SH + col0 + t * 16 + n] = (bf16_t)gel;
      }
  }
  __syncthreads();

  // ---- Stage F: out = LN2(x + H @ W2^T + b2) -> global ----
  {
    const int col0 = h * 256;
#pragma unroll
    for (int t = 0; t < 16; ++t) {
      float bz = b2[col0 + t * 16 + n];
#pragma unroll
      for (int r = 0; r < 8; ++r) acc[t][r] = bz;
    }
    gemm_staged<32, 32768, 2048>(acc, R0 + strip * 16 * SH, SH,
                                 wpkB + (unsigned long long)W2_OFF * 2, PNL, h, lane, wid);

    float sum[8], sq[8];
#pragma unroll
    for (int r = 0; r < 8; ++r) { sum[r] = 0.f; sq[r] = 0.f; }
#pragma unroll
    for (int t = 0; t < 16; ++t)
#pragma unroll
      for (int r = 0; r < 8; ++r) {
        int row = strip * 16 + khalf * 8 + r;
        float v = acc[t][r] + (float)R1[row * SD + col0 + t * 16 + n];
        acc[t][r] = v; sum[r] += v; sq[r] += v * v;
      }
#pragma unroll
    for (int r = 0; r < 8; ++r)
      for (int msk = 1; msk < 16; msk <<= 1) {
        sum[r] += __shfl_xor(sum[r], msk, 32);
        sq[r]  += __shfl_xor(sq[r],  msk, 32);
      }
    if (n == 0)
#pragma unroll
      for (int r = 0; r < 8; ++r) {
        int row = strip * 16 + khalf * 8 + r;
        red[row * 2 + h] = sum[r];
        red[128 + row * 2 + h] = sq[r];
      }
    __syncthreads();
    float mu[8], rstd[8];
#pragma unroll
    for (int r = 0; r < 8; ++r) {
      int row = strip * 16 + khalf * 8 + r;
      float s = red[row * 2] + red[row * 2 + 1];
      float q = red[128 + row * 2] + red[128 + row * 2 + 1];
      float m_ = s * (1.0f / 512.0f);
      mu[r] = m_;
      rstd[r] = rsqrtf(q * (1.0f / 512.0f) - m_ * m_ + EPSV);
    }
#pragma unroll
    for (int t = 0; t < 16; ++t) {
      int col = col0 + t * 16 + n;
      float gg = g2[col], bb = be2[col];
#pragma unroll
      for (int r = 0; r < 8; ++r) {
        int grow = rowbase + strip * 16 + khalf * 8 + r;
        out[(size_t)grow * DD + col] = (acc[t][r] - mu[r]) * rstd[r] * gg + bb;
      }
    }
  }
}

extern "C" void kernel_launch(void* const* d_in, const int* in_sizes, int n_in,
                              void* d_out, int out_size, void* d_ws, size_t ws_size,
                              hipStream_t stream) {
  (void)in_sizes; (void)n_in; (void)out_size; (void)ws_size;
  const float* cell = (const float*)d_in[0];
  const float* pert = (const float*)d_in[1];
  const float* Wp   = (const float*)d_in[2];
  const float* bp   = (const float*)d_in[3];
  // d_in[4..7] = Wq,bq,Wk,bk : dead (softmax over one key == 1)
  const float* Wv   = (const float*)d_in[8];
  const float* bv   = (const float*)d_in[9];
  const float* Wo   = (const float*)d_in[10];
  const float* bo   = (const float*)d_in[11];
  const float* g1   = (const float*)d_in[12];
  const float* be1  = (const float*)d_in[13];
  const float* g2   = (const float*)d_in[14];
  const float* be2  = (const float*)d_in[15];
  const float* W1   = (const float*)d_in[16];
  const float* b1   = (const float*)d_in[17];
  const float* W2   = (const float*)d_in[18];
  const float* b2   = (const float*)d_in[19];
  bf16_t* wpk = (bf16_t*)d_ws;

  // Pack weights into WMMA-B fragment order (bf16) in workspace: 3.25 MB, L2-resident.
  pack_w<<<256, 256, 0, stream>>>(Wp, wpk + WP_OFF, 512, 256);
  pack_w<<<256, 256, 0, stream>>>(Wv, wpk + WV_OFF, 512, 512);
  pack_w<<<256, 256, 0, stream>>>(Wo, wpk + WO_OFF, 512, 512);
  pack_w<<<512, 256, 0, stream>>>(W1, wpk + W1_OFF, 1024, 512);
  pack_w<<<512, 256, 0, stream>>>(W2, wpk + W2_OFF, 512, 1024);

  static_assert(SMEM_BYTES <= 320 * 1024, "LDS over WGP budget");
  hipFuncSetAttribute((const void*)fused_block,
                      hipFuncAttributeMaxDynamicSharedMemorySize, SMEM_BYTES);
  fused_block<<<BB / BM, 256, SMEM_BYTES, stream>>>(cell, pert, wpk, bp, bv, bo, b1, b2,
                                                    g1, be1, g2, be2, (float*)d_out);
}
